// MinibatchKmeansClusterOneStep_37091337568421
// MI455X (gfx1250) — matrix operand order repaired and checked by reading, not compiled, not run on hardware
//
#include <hip/hip_runtime.h>
#include <hip/hip_bf16.h>
#include <math.h>

// Problem constants (G=1)
constexpr int N_ = 32768;
constexpr int K_ = 2048;
constexpr int D_ = 512;

typedef __bf16 bf16;
typedef __attribute__((ext_vector_type(16))) __bf16 v16bf;
typedef __attribute__((ext_vector_type(8)))  float  v8f;

// ---------------------------------------------------------------------------
// fp32 -> bf16 hi/lo split (truncating): x ~= hi + lo, |err| ~ 2^-16 * |x|
// ---------------------------------------------------------------------------
__device__ __forceinline__ void split_bf16(float x, unsigned short& h, unsigned short& l) {
    unsigned u  = __builtin_bit_cast(unsigned, x);
    unsigned hu = u & 0xFFFF0000u;
    h = (unsigned short)(hu >> 16);
    float r = x - __builtin_bit_cast(float, hu);
    unsigned ru = __builtin_bit_cast(unsigned, r);
    l = (unsigned short)(ru >> 16);
}

union AFrag { v16bf v; unsigned short u[16]; };

// Build one 16x32 A fragment (contiguous k order, same as B) from an fp32 sample row.
//   lane half 0: elems 0..15 -> K=0..15
//   lane half 1: elems 0..15 -> K=16..31
__device__ __forceinline__ void build_a(const float* __restrict__ rowp, int d0, int akb,
                                        AFrag& ah, AFrag& al) {
    const float4* p0 = (const float4*)(rowp + d0 + akb);        // K = akb..akb+7
    const float4* p1 = (const float4*)(rowp + d0 + 8 + akb);    // K = 8+akb..
    float4 x0 = p0[0], x1 = p0[1], y0 = p1[0], y1 = p1[1];
    float xs[16] = {x0.x, x0.y, x0.z, x0.w, x1.x, x1.y, x1.z, x1.w,
                    y0.x, y0.y, y0.z, y0.w, y1.x, y1.y, y1.z, y1.w};
#pragma unroll
    for (int e = 0; e < 16; ++e) split_bf16(xs[e], ah.u[e], al.u[e]);
}

// ---------------------------------------------------------------------------
// Zero the accumulator region of the workspace (csum, ccnt, inertia pad)
// ---------------------------------------------------------------------------
__global__ void zero_ws_kernel(float* p, int count) {
    int i = blockIdx.x * blockDim.x + threadIdx.x;
    if (i < count) p[i] = 0.0f;
}

// ---------------------------------------------------------------------------
// Means prep: bf16 hi/lo split + ||m||^2.  One block per cluster.
// ---------------------------------------------------------------------------
__global__ __launch_bounds__(256)
void prep_means_kernel(const float* __restrict__ means,
                       unsigned short* __restrict__ mh,
                       unsigned short* __restrict__ ml,
                       float* __restrict__ mnorm) {
    int k = blockIdx.x;
    int t = threadIdx.x;
    float acc = 0.0f;
    for (int d = t; d < D_; d += 256) {
        float x = means[(size_t)k * D_ + d];
        unsigned short h, l;
        split_bf16(x, h, l);
        mh[(size_t)k * D_ + d] = h;
        ml[(size_t)k * D_ + d] = l;
        acc += x * x;
    }
    __shared__ float red[256];
    red[t] = acc;
    __syncthreads();
    for (int s = 128; s > 0; s >>= 1) {
        if (t < s) red[t] += red[t + s];
        __syncthreads();
    }
    if (t == 0) mnorm[k] = red[0];
}

// ---------------------------------------------------------------------------
// Sample norms ||s||^2.  One block per sample.
// ---------------------------------------------------------------------------
__global__ __launch_bounds__(256)
void snorm_kernel(const float* __restrict__ samples, float* __restrict__ snorm) {
    int n = blockIdx.x;
    int t = threadIdx.x;
    const float* s = samples + (size_t)n * D_;
    float acc = 0.0f;
    for (int d = t; d < D_; d += 256) { float x = s[d]; acc += x * x; }
    __shared__ float red[256];
    red[t] = acc;
    __syncthreads();
    for (int sft = 128; sft > 0; sft >>= 1) {
        if (t < sft) red[t] += red[t + sft];
        __syncthreads();
    }
    if (t == 0) snorm[n] = red[0];
}

// ---------------------------------------------------------------------------
// Assignment kernel: per block (128 threads = 4 waves), 128 samples x all
// 2048 clusters.  Each wave owns TWO 16-row M-tiles (32 samples) so every
// B fragment loaded from LDS feeds 6 WMMAs.  Per pass: 64 clusters
// (4 n-tiles); dot accumulated over D in 32-wide steps with 3x
// v_wmma_f32_16x16x32_bf16 per (n-tile, m-tile): hi*hi + hi*lo + lo*hi.
// Means tiles double-buffered in LDS (one barrier per D-step); next cluster
// group prefetched via global_prefetch.  Running per-slot argmin; final
// cross-lane argmin via shfl_xor within each 16-lane half (wave32).
// ---------------------------------------------------------------------------
__global__ __launch_bounds__(128)
void assign_kernel(const float* __restrict__ samples,
                   const unsigned short* __restrict__ mh,
                   const unsigned short* __restrict__ ml,
                   const float* __restrict__ snorm,
                   const float* __restrict__ mnorm,
                   int* __restrict__ bins,
                   float* __restrict__ mind) {
    constexpr int NG = K_ / 64;        // 32 cluster groups
    constexpr int ND = D_ / 32;        // 16 d-steps
    constexpr int S  = NG * ND;        // 512 staged tiles

    __shared__ __align__(32) unsigned short lds_h[2][64 * 32];
    __shared__ __align__(32) unsigned short lds_l[2][64 * 32];

    const int t    = threadIdx.x;
    const int wave = t >> 5;
    const int lane = t & 31;
    const int lh   = lane >> 4;   // lane half (0/1)
    const int ln   = lane & 15;

    const int m0 = blockIdx.x * 128 + wave * 32;   // 32 rows per wave

    const float* aptr0 = samples + (size_t)(m0 + ln) * D_;
    const float* aptr1 = samples + (size_t)(m0 + 16 + ln) * D_;
    const int akb = lh * 16;

    // ||s||^2 per accumulator slot: slot (mt, v) holds M = mt*16 + v + 8*lh
    float sn[2][8];
#pragma unroll
    for (int mt = 0; mt < 2; ++mt)
#pragma unroll
        for (int v = 0; v < 8; ++v) sn[mt][v] = snorm[m0 + mt * 16 + v + 8 * lh];

    float minv[2][8];
    int   mini[2][8];
#pragma unroll
    for (int mt = 0; mt < 2; ++mt)
#pragma unroll
        for (int v = 0; v < 8; ++v) { minv[mt][v] = 3.4e38f; mini[mt][v] = 0; }

    // cooperative B staging: 128 threads x 16 shorts (2 x uint4) per array
    const int bc = t >> 1;          // cluster 0..63 within group
    const int bd = (t & 1) * 16;    // d segment (16 shorts)

    auto stage = [&](int b, int s) {
        const int ngs = s >> 4;
        const int dss = s & 15;
        const size_t roff = (size_t)(ngs * 64 + bc) * D_ + dss * 32 + bd;
        *(uint4*)&lds_h[b][bc * 32 + bd]     = *(const uint4*)&mh[roff];
        *(uint4*)&lds_h[b][bc * 32 + bd + 8] = *(const uint4*)&mh[roff + 8];
        *(uint4*)&lds_l[b][bc * 32 + bd]     = *(const uint4*)&ml[roff];
        *(uint4*)&lds_l[b][bc * 32 + bd + 8] = *(const uint4*)&ml[roff + 8];
    };

    v8f acc[4][2];

    stage(0, 0);
    __syncthreads();

    for (int s = 0; s < S; ++s) {
        const int ng  = s >> 4;
        const int ds  = s & 15;
        const int cur = s & 1;

        if (s + 1 < S) stage(cur ^ 1, s + 1);

        if (ds == 0) {
            // reset accumulators; prefetch next cluster group's means rows
#pragma unroll
            for (int tt = 0; tt < 4; ++tt)
#pragma unroll
                for (int mt = 0; mt < 2; ++mt) acc[tt][mt] = (v8f){};
            if (ng + 1 < NG) {
                __builtin_prefetch(mh + (size_t)((ng + 1) * 64 + bc) * D_ + bd, 0, 1);
                __builtin_prefetch(ml + (size_t)((ng + 1) * 64 + bc) * D_ + bd, 0, 1);
            }
        }

        const int d0 = ds * 32;
        AFrag ah0, al0, ah1, al1;
        build_a(aptr0, d0, akb, ah0, al0);
        build_a(aptr1, d0, akb, ah1, al1);

        // 4 n-tiles of 16 clusters; B layout: elem e -> K = 16*lh + e, col = ln
#pragma unroll
        for (int tt = 0; tt < 4; ++tt) {
            const int c = tt * 16 + ln;
            v16bf bh = *(const v16bf*)&lds_h[cur][c * 32 + lh * 16];
            v16bf bl = *(const v16bf*)&lds_l[cur][c * 32 + lh * 16];
            acc[tt][0] = __builtin_amdgcn_wmma_f32_16x16x32_bf16(
                false, ah0.v, false, bh, (short)0, acc[tt][0], false, false);
            acc[tt][0] = __builtin_amdgcn_wmma_f32_16x16x32_bf16(
                false, ah0.v, false, bl, (short)0, acc[tt][0], false, false);
            acc[tt][0] = __builtin_amdgcn_wmma_f32_16x16x32_bf16(
                false, al0.v, false, bh, (short)0, acc[tt][0], false, false);
            acc[tt][1] = __builtin_amdgcn_wmma_f32_16x16x32_bf16(
                false, ah1.v, false, bh, (short)0, acc[tt][1], false, false);
            acc[tt][1] = __builtin_amdgcn_wmma_f32_16x16x32_bf16(
                false, ah1.v, false, bl, (short)0, acc[tt][1], false, false);
            acc[tt][1] = __builtin_amdgcn_wmma_f32_16x16x32_bf16(
                false, al1.v, false, bh, (short)0, acc[tt][1], false, false);
        }

        if (ds == 15) {
            // fold this 64-cluster group into the running argmin
            const int kbase = ng * 64;
#pragma unroll
            for (int tt = 0; tt < 4; ++tt) {
                const int c = kbase + tt * 16 + ln;
                const float mn = mnorm[c];
#pragma unroll
                for (int mt = 0; mt < 2; ++mt)
#pragma unroll
                    for (int v = 0; v < 8; ++v) {
                        const float dist = sn[mt][v] + mn - 2.0f * acc[tt][mt][v];
                        if (dist < minv[mt][v]) { minv[mt][v] = dist; mini[mt][v] = c; }
                    }
            }
        }

        __syncthreads();
    }

    // cross-lane argmin inside each 16-lane half (wave32; bit4 untouched)
#pragma unroll
    for (int mt = 0; mt < 2; ++mt)
#pragma unroll
        for (int v = 0; v < 8; ++v) {
            float mv = minv[mt][v];
            int   mi = mini[mt][v];
#pragma unroll
            for (int off = 1; off < 16; off <<= 1) {
                float ov = __shfl_xor(mv, off, 32);
                int   oi = __shfl_xor(mi, off, 32);
                if (ov < mv || (ov == mv && oi < mi)) { mv = ov; mi = oi; }
            }
            if (ln == 0) {
                const int row = m0 + mt * 16 + v + 8 * lh;
                bins[row] = mi;
                mind[row] = mv > 0.0f ? mv : 0.0f;
            }
        }
}

// ---------------------------------------------------------------------------
// Scatter: per-cluster counts, per-cluster sample sums, inertia.
// One block per sample; avg 16 samples/cluster -> low atomic contention.
// ---------------------------------------------------------------------------
__global__ __launch_bounds__(256)
void scatter_kernel(const float* __restrict__ samples,
                    const int* __restrict__ bins,
                    const float* __restrict__ mind,
                    float* __restrict__ csum,
                    float* __restrict__ ccnt,
                    float* __restrict__ inert) {
    const int n = blockIdx.x;
    const int k = bins[n];
    const float* s = samples + (size_t)n * D_;
    float* dst = csum + (size_t)k * D_;
    for (int d = threadIdx.x; d < D_; d += blockDim.x)
        atomicAdd(&dst[d], s[d]);
    if (threadIdx.x == 0) {
        atomicAdd(&ccnt[k], 1.0f);
        atomicAdd(inert, sqrtf(fmaxf(mind[n], 0.0f)));
    }
}

// ---------------------------------------------------------------------------
// Finalize: new_means = (csum + means*wsum) / (wsum + cnt), keep old mean if
// cnt==0; append inertia as last output element.
// ---------------------------------------------------------------------------
__global__ __launch_bounds__(256)
void finalize_kernel(const float* __restrict__ means,
                     const float* __restrict__ wsum,
                     const float* __restrict__ csum,
                     const float* __restrict__ ccnt,
                     const float* __restrict__ inert,
                     float* __restrict__ out) {
    const int i = blockIdx.x * blockDim.x + threadIdx.x;
    if (i < K_ * D_) {
        const int k = i / D_;
        const float w = wsum[k];
        const float c = ccnt[k];
        const float nws = w + c;
        const float alpha = 1.0f / (nws == 0.0f ? 1.0f : nws);
        const float val = (csum[i] + means[i] * w) * alpha;
        out[i] = (c == 0.0f) ? means[i] : val;
    }
    if (i == 0) out[K_ * D_] = *inert;
}

// ---------------------------------------------------------------------------
// Launch
// ---------------------------------------------------------------------------
extern "C" void kernel_launch(void* const* d_in, const int* in_sizes, int n_in,
                              void* d_out, int out_size, void* d_ws, size_t ws_size,
                              hipStream_t stream) {
    const float* samples = (const float*)d_in[0];   // [N, D]
    const float* means   = (const float*)d_in[1];   // [K, D]
    const float* wsum    = (const float*)d_in[2];   // [K]
    float* out = (float*)d_out;                     // [K*D + 1]

    // Workspace layout (all offsets 64B-aligned)
    char* ws = (char*)d_ws;
    size_t off = 0;
    float* csum  = (float*)(ws + off); off += (size_t)K_ * D_ * sizeof(float);   // 4 MB
    float* ccnt  = (float*)(ws + off); off += (size_t)K_ * sizeof(float);        // 8 KB
    float* inert = (float*)(ws + off); off += 64;                                 // pad
    float* mnorm = (float*)(ws + off); off += (size_t)K_ * sizeof(float);
    float* snorm = (float*)(ws + off); off += (size_t)N_ * sizeof(float);
    int*   bins  = (int*)  (ws + off); off += (size_t)N_ * sizeof(int);
    float* mind  = (float*)(ws + off); off += (size_t)N_ * sizeof(float);
    unsigned short* mh = (unsigned short*)(ws + off); off += (size_t)K_ * D_ * sizeof(unsigned short);
    unsigned short* ml = (unsigned short*)(ws + off); off += (size_t)K_ * D_ * sizeof(unsigned short);

    // 1) zero accumulators (csum, ccnt, inertia pad are contiguous)
    const int zcount = K_ * D_ + K_ + 16;
    zero_ws_kernel<<<(zcount + 255) / 256, 256, 0, stream>>>(csum, zcount);

    // 2) means: bf16 hi/lo split + norms
    prep_means_kernel<<<K_, 256, 0, stream>>>(means, mh, ml, mnorm);

    // 3) sample norms
    snorm_kernel<<<N_, 256, 0, stream>>>(samples, snorm);

    // 4) WMMA distance + argmin (128 threads, 128 samples per block)
    assign_kernel<<<N_ / 128, 128, 0, stream>>>(samples, mh, ml, snorm, mnorm, bins, mind);

    // 5) scatter sums / counts / inertia
    scatter_kernel<<<N_, 256, 0, stream>>>(samples, bins, mind, csum, ccnt, inert);

    // 6) finalize new means + inertia
    finalize_kernel<<<(K_ * D_ + 255) / 256, 256, 0, stream>>>(means, wsum, csum, ccnt, inert, out);
}
